// MultiHeadAttention_36507222016671
// MI455X (gfx1250) — matrix operand order repaired and checked by reading, not compiled, not run on hardware
//
#include <hip/hip_runtime.h>
#include <hip/hip_bf16.h>

// ---------------------------------------------------------------------------
// CDNA5 (gfx1250) fused multi-head cosine attention, bf16 WMMA pipeline.
// B operands staged in LDS transposed ([n][k]) -> all fragment loads are
// two contiguous 16B runs per lane (2x ds_load_b128). K/V tiles fetched with the
// gfx1250 async global->LDS data mover (ASYNCcnt).
// ---------------------------------------------------------------------------

typedef __bf16 bf16_t;
typedef __attribute__((ext_vector_type(16))) __bf16 v16bf;
typedef __attribute__((ext_vector_type(8)))  __bf16 v8bf;
typedef __attribute__((ext_vector_type(4)))  __bf16 v4bf;
typedef __attribute__((ext_vector_type(8)))  float  v8f;

#define BS   2
#define SEQ  2048
#define DIM  1024
#define NH   16
#define DPH  64
#define MTOT (BS * SEQ)
#define LDP  72   // bf16 LDS row pitch: 144B = 9*16 keeps b128 alignment
#define LDF  68   // f32 LDS row pitch: 272B = 17*16 keeps float4 alignment

__device__ __forceinline__ v8f vzero8() {
    v8f z;
#pragma unroll
    for (int i = 0; i < 8; ++i) z[i] = 0.0f;
    return z;
}

// A-fragment (16x32 bf16, MxK) from row-major [m][k] LDS tile (ISA 7.12.2):
// per lane two contiguous 8-elem runs -> 2x ds_load_b128.
__device__ __forceinline__ v16bf load_afrag(const bf16_t* p, int ld, int lane) {
    const int half = lane >> 4, r = lane & 15;
    v16bf a;
#pragma unroll
    for (int v = 0; v < 8; ++v) {
        const int kb = ((v < 4) ? (2 * v) : (16 + 2 * (v - 4))) + 8 * half;
        a[2 * v]     = p[r * ld + kb];
        a[2 * v + 1] = p[r * ld + kb + 1];
    }
    return a;
}

// B-fragment (32x16 bf16, KxN) from TRANSPOSED [n][k] LDS tile:
// element j <-> k = 8*half + j (j<8), 16 + 8*half + (j-8) (j>=8) -> 2x ds_load_b128.
__device__ __forceinline__ v16bf load_bfragT(const bf16_t* p, int ld, int lane) {
    const int half = lane >> 4, n = lane & 15;
    v16bf b;
#pragma unroll
    for (int j = 0; j < 16; ++j)
        b[j] = p[n * ld + ((j < 8) ? j : (j + 8)) + 8 * half];
    return b;
}

__device__ __forceinline__ v8f wmma_bf16(v16bf a, v16bf b, v8f c) {
    return __builtin_amdgcn_wmma_f32_16x16x32_bf16(
        false, a, false, b, (short)0, c, false, false);
}

// gfx1250 async global->LDS 16B copy (GV mode, null saddr). LDS address is
// the low 32 bits of the flat pointer (ISA 10.2 aperture mapping).
__device__ __forceinline__ void async_copy16(void* lds_dst, const void* gsrc) {
    const unsigned ldsa = (unsigned)(unsigned long long)lds_dst;
    asm volatile("global_load_async_to_lds_b128 %0, %1, off"
                 :: "v"(ldsa), "v"(gsrc) : "memory");
}
__device__ __forceinline__ void async_wait0() {
    asm volatile("s_wait_asynccnt 0" ::: "memory");
}

// ---------------------------------------------------------------------------
// GEMM: 64x64 tile of C = A[M x 1024] * W[1024 x 1024] + bias, epilogues:
//   MODE 0 (Q): per-row L2 norm over head tile, * scale, bf16 [b,h,s,d]
//   MODE 1 (K): per-row L2 norm, bf16 [b,h,s,d]
//   MODE 2 (V): bias only, bf16 transposed [b,h,d,s]
//   MODE 3 (O): bias only, fp32 [m, n]  (final output)
// block = 128 threads (4 waves, 2x2), grid = (DIM/64, MTOT/64)
// ---------------------------------------------------------------------------
template <int MODE>
__global__ __launch_bounds__(128) void gemm64(
    const float* __restrict__ A, const float* __restrict__ W,
    const float* __restrict__ bias, void* __restrict__ out,
    const float* __restrict__ scale_p)
{
    __shared__ bf16_t As[64][LDP];   // [m][k], k-tile = 64
    __shared__ bf16_t Bt[64][LDP];   // transposed: [n][k]
    __shared__ float  Cs[64][LDF];

    const int bx = blockIdx.x;      // n tile == head index
    const int by = blockIdx.y;      // m tile
    const int tid  = threadIdx.x;
    const int lane = tid & 31;
    const int wid  = tid >> 5;
    const int m_off = (wid >> 1) * 32;
    const int n_off = (wid & 1) * 32;

    v8f acc[2][2];
#pragma unroll
    for (int i = 0; i < 2; ++i)
#pragma unroll
        for (int j = 0; j < 2; ++j) acc[i][j] = vzero8();

    for (int k0 = 0; k0 < DIM; k0 += 64) {
        // A tile 64x64 fp32 -> bf16, 512 8-elem groups / 128 threads
#pragma unroll
        for (int i = 0; i < 4; ++i) {
            const int idx = tid + i * 128;
            const int row = idx >> 3;
            const int col = (idx & 7) * 8;
            const float* src = &A[(size_t)(by * 64 + row) * DIM + k0 + col];
            const float4 f0 = *(const float4*)src;
            const float4 f1 = *(const float4*)(src + 4);
            v8bf pk;
            pk[0] = (bf16_t)f0.x; pk[1] = (bf16_t)f0.y;
            pk[2] = (bf16_t)f0.z; pk[3] = (bf16_t)f0.w;
            pk[4] = (bf16_t)f1.x; pk[5] = (bf16_t)f1.y;
            pk[6] = (bf16_t)f1.z; pk[7] = (bf16_t)f1.w;
            *(v8bf*)&As[row][col] = pk;
        }
        // W tile 64x64, transposed via 4x4 register blocks (2 per thread)
#pragma unroll
        for (int i = 0; i < 2; ++i) {
            const int bi = tid + i * 128;       // 0..255
            const int nb = (bi >> 4) * 4;       // n block base
            const int kb = (bi & 15) * 4;       // k block base
            const float* src = &W[(size_t)(k0 + kb) * DIM + bx * 64 + nb];
            const float4 r0 = *(const float4*)(src);
            const float4 r1 = *(const float4*)(src + DIM);
            const float4 r2 = *(const float4*)(src + 2 * DIM);
            const float4 r3 = *(const float4*)(src + 3 * DIM);
            const v4bf c0 = {(bf16_t)r0.x, (bf16_t)r1.x, (bf16_t)r2.x, (bf16_t)r3.x};
            const v4bf c1 = {(bf16_t)r0.y, (bf16_t)r1.y, (bf16_t)r2.y, (bf16_t)r3.y};
            const v4bf c2 = {(bf16_t)r0.z, (bf16_t)r1.z, (bf16_t)r2.z, (bf16_t)r3.z};
            const v4bf c3 = {(bf16_t)r0.w, (bf16_t)r1.w, (bf16_t)r2.w, (bf16_t)r3.w};
            *(v4bf*)&Bt[nb + 0][kb] = c0;
            *(v4bf*)&Bt[nb + 1][kb] = c1;
            *(v4bf*)&Bt[nb + 2][kb] = c2;
            *(v4bf*)&Bt[nb + 3][kb] = c3;
        }
        __syncthreads();

#pragma unroll
        for (int kk = 0; kk < 64; kk += 32) {
            const v16bf a0 = load_afrag(&As[m_off][kk], LDP, lane);
            const v16bf a1 = load_afrag(&As[m_off + 16][kk], LDP, lane);
            const v16bf b0 = load_bfragT(&Bt[n_off][kk], LDP, lane);
            const v16bf b1 = load_bfragT(&Bt[n_off + 16][kk], LDP, lane);
            acc[0][0] = wmma_bf16(a0, b0, acc[0][0]);
            acc[0][1] = wmma_bf16(a0, b1, acc[0][1]);
            acc[1][0] = wmma_bf16(a1, b0, acc[1][0]);
            acc[1][1] = wmma_bf16(a1, b1, acc[1][1]);
        }
        __syncthreads();
    }

    // stage accumulators (+bias) to LDS: row = r + 8*half, col = lane&15
    const int half = lane >> 4, nl = lane & 15;
    const float bsv0 = bias[bx * 64 + n_off + nl];
    const float bsv1 = bias[bx * 64 + n_off + 16 + nl];
#pragma unroll
    for (int ms = 0; ms < 2; ++ms)
#pragma unroll
        for (int ns = 0; ns < 2; ++ns)
#pragma unroll
            for (int r = 0; r < 8; ++r) {
                const int mr = m_off + 16 * ms + 8 * half + r;
                const int nc = n_off + 16 * ns + nl;
                Cs[mr][nc] = acc[ms][ns][r] + (ns ? bsv1 : bsv0);
            }
    __syncthreads();

    if (MODE == 0 || MODE == 1) {
        if (tid < 64) {
            float ss = 0.0f;
#pragma unroll
            for (int j4 = 0; j4 < 16; ++j4) {
                const float4 v = *(const float4*)&Cs[tid][j4 * 4];
                ss += v.x * v.x + v.y * v.y + v.z * v.z + v.w * v.w;
            }
            float inv = 1.0f / fmaxf(sqrtf(ss), 1e-12f);
            if (MODE == 0) inv *= *scale_p;
#pragma unroll
            for (int j4 = 0; j4 < 16; ++j4) {
                float4 v = *(const float4*)&Cs[tid][j4 * 4];
                v.x *= inv; v.y *= inv; v.z *= inv; v.w *= inv;
                *(float4*)&Cs[tid][j4 * 4] = v;
            }
        }
        __syncthreads();
    }

    // vectorized write-out
    if (MODE == 3) {
#pragma unroll
        for (int i = 0; i < 8; ++i) {
            const int idx = tid + i * 128;          // 1024 float4 groups
            const int mr = idx >> 4, nc = (idx & 15) * 4;
            const int grow = by * 64 + mr;
            *(float4*)&((float*)out)[(size_t)grow * DIM + bx * 64 + nc] =
                *(const float4*)&Cs[mr][nc];
        }
    } else if (MODE == 2) {                         // V: [b,h,d,s], s contiguous
#pragma unroll
        for (int i = 0; i < 8; ++i) {
            const int idx = tid + i * 128;          // 1024 groups: (nc, 4 rows)
            const int nc = idx & 63;
            const int mr = (idx >> 6) * 4;
            const int grow = by * 64 + mr;
            const int b = grow >> 11, s = grow & (SEQ - 1);
            const v4bf pk = {(bf16_t)Cs[mr + 0][nc], (bf16_t)Cs[mr + 1][nc],
                             (bf16_t)Cs[mr + 2][nc], (bf16_t)Cs[mr + 3][nc]};
            *(v4bf*)&((bf16_t*)out)[(((size_t)b * NH + bx) * DPH + nc) * SEQ + s] = pk;
        }
    } else {                                        // Q,K: [b,h,s,d], d contiguous
#pragma unroll
        for (int i = 0; i < 8; ++i) {
            const int idx = tid + i * 128;
            const int mr = idx >> 4, nc = (idx & 15) * 4;
            const int grow = by * 64 + mr;
            const int b = grow >> 11, s = grow & (SEQ - 1);
            const float4 v = *(const float4*)&Cs[mr][nc];
            const v4bf pk = {(bf16_t)v.x, (bf16_t)v.y, (bf16_t)v.z, (bf16_t)v.w};
            *(v4bf*)&((bf16_t*)out)[(((size_t)b * NH + bx) * SEQ + s) * DPH + nc] = pk;
        }
    }
}

// ---------------------------------------------------------------------------
// Flash-style attention: block = (64 q-rows, one (b,h)); online softmax over
// 64-key tiles. Q,K global [b,h,s,d]; V global [b,h,d,s].
// K/V tiles fetched via async global->LDS (ASYNCcnt). Mask folded into an
// additive bias applied when staging scores.
// ---------------------------------------------------------------------------
__global__ __launch_bounds__(128) void attn_kernel(
    const bf16_t* __restrict__ Q, const bf16_t* __restrict__ K,
    const bf16_t* __restrict__ V, const int* __restrict__ mask,
    float* __restrict__ ctx)
{
    __shared__ bf16_t Qs[64][LDP];   // [q][d]
    __shared__ bf16_t Ks[64][LDP];   // [key][d]  == B^T for S-gemm
    __shared__ bf16_t Vt[64][LDP];   // [d][key]  == B^T for O-gemm
    __shared__ bf16_t Ps[64][LDP];   // [q][key]  A operand for O-gemm
    __shared__ float  Ss[64][LDF];
    __shared__ float  Mrow[64], Lrow[64], Arow[64], MB[64];

    const int q0 = blockIdx.x * 64;
    const int bh = blockIdx.y;              // b*NH + h
    const int b  = bh / NH;
    const int h  = bh - b * NH;
    const int tid  = threadIdx.x;
    const int lane = tid & 31;
    const int wid  = tid >> 5;
    const int m_off = (wid >> 1) * 32;
    const int n_off = (wid & 1) * 32;
    const int half = lane >> 4, nl = lane & 15;
    const float NEG_INF = -__builtin_inff();

    const bf16_t* Qbase = Q + ((size_t)bh * SEQ + q0) * DPH;
    const bf16_t* Kbase = K + (size_t)bh * SEQ * DPH;   // [s][d]
    const bf16_t* Vbase = V + (size_t)bh * DPH * SEQ;   // [d][s]

    // Q tile: 512 8-elem (16B) groups / 128 threads
#pragma unroll
    for (int i = 0; i < 4; ++i) {
        const int idx = tid + i * 128;
        const int row = idx >> 3, col = (idx & 7) * 8;
        *(uint4*)&Qs[row][col] = *(const uint4*)&Qbase[(size_t)row * DPH + col];
    }
    if (tid < 64) { Mrow[tid] = NEG_INF; Lrow[tid] = 0.0f; }

    v8f o[2][2];
#pragma unroll
    for (int i = 0; i < 2; ++i)
#pragma unroll
        for (int j = 0; j < 2; ++j) o[i][j] = vzero8();
    __syncthreads();

    for (int kb = 0; kb < SEQ; kb += 64) {
        // async K tile [key][d] and V^T tile [d][key] (contiguous rows)
#pragma unroll
        for (int i = 0; i < 4; ++i) {
            const int idx = tid + i * 128;
            const int row = idx >> 3, col = (idx & 7) * 8;
            async_copy16(&Ks[row][col], &Kbase[(size_t)(kb + row) * DPH + col]);
            async_copy16(&Vt[row][col], &Vbase[(size_t)row * SEQ + kb + col]);
        }
        if (tid < 64) MB[tid] = mask[b * SEQ + kb + tid] ? 0.0f : NEG_INF;
        async_wait0();
        __syncthreads();

        // S = Q * K^T  (64x64), contraction over d in two 32-steps
        v8f s[2][2];
#pragma unroll
        for (int i = 0; i < 2; ++i)
#pragma unroll
            for (int j = 0; j < 2; ++j) s[i][j] = vzero8();
#pragma unroll
        for (int kk = 0; kk < 64; kk += 32) {
            const v16bf a0 = load_afrag(&Qs[m_off][kk], LDP, lane);
            const v16bf a1 = load_afrag(&Qs[m_off + 16][kk], LDP, lane);
            const v16bf b0 = load_bfragT(&Ks[n_off][kk], LDP, lane);
            const v16bf b1 = load_bfragT(&Ks[n_off + 16][kk], LDP, lane);
            s[0][0] = wmma_bf16(a0, b0, s[0][0]);
            s[0][1] = wmma_bf16(a0, b1, s[0][1]);
            s[1][0] = wmma_bf16(a1, b0, s[1][0]);
            s[1][1] = wmma_bf16(a1, b1, s[1][1]);
        }
        // stage scores (+mask bias) to LDS
        const float mb0 = MB[n_off + nl];
        const float mb1 = MB[n_off + 16 + nl];
#pragma unroll
        for (int ms = 0; ms < 2; ++ms)
#pragma unroll
            for (int ns = 0; ns < 2; ++ns)
#pragma unroll
                for (int r = 0; r < 8; ++r)
                    Ss[m_off + 16 * ms + 8 * half + r][n_off + 16 * ns + nl] =
                        s[ms][ns][r] + (ns ? mb1 : mb0);
        __syncthreads();

        // online softmax, one row per thread (tid < 64), vectorized scans
        if (tid < 64) {
            const float mprev = Mrow[tid];
            float mloc = NEG_INF;
#pragma unroll
            for (int c4 = 0; c4 < 16; ++c4) {
                const float4 s4 = *(const float4*)&Ss[tid][c4 * 4];
                mloc = fmaxf(mloc, fmaxf(fmaxf(s4.x, s4.y), fmaxf(s4.z, s4.w)));
            }
            const float mnew = fmaxf(mprev, mloc);
            const bool finite = (mnew > -3.0e38f);
            const float al = finite ? __expf(mprev - mnew) : 0.0f;
            float lsum = 0.0f;
#pragma unroll
            for (int c4 = 0; c4 < 16; ++c4) {
                const float4 s4 = *(const float4*)&Ss[tid][c4 * 4];
                const float p0 = finite ? __expf(s4.x - mnew) : 0.0f;
                const float p1 = finite ? __expf(s4.y - mnew) : 0.0f;
                const float p2 = finite ? __expf(s4.z - mnew) : 0.0f;
                const float p3 = finite ? __expf(s4.w - mnew) : 0.0f;
                const v4bf pk = {(bf16_t)p0, (bf16_t)p1, (bf16_t)p2, (bf16_t)p3};
                *(v4bf*)&Ps[tid][c4 * 4] = pk;
                lsum += (p0 + p1) + (p2 + p3);
            }
            Lrow[tid] = Lrow[tid] * al + lsum;
            Mrow[tid] = mnew;
            Arow[tid] = al;
        }
        __syncthreads();

        // rescale running O by alpha (vectorized row reads), then O += P * V
#pragma unroll
        for (int ms = 0; ms < 2; ++ms) {
            const int base = m_off + 16 * ms + 8 * half;
            const float4 a0 = *(const float4*)&Arow[base];
            const float4 a1 = *(const float4*)&Arow[base + 4];
            const float av[8] = {a0.x, a0.y, a0.z, a0.w, a1.x, a1.y, a1.z, a1.w};
#pragma unroll
            for (int r = 0; r < 8; ++r) {
                o[ms][0][r] *= av[r];
                o[ms][1][r] *= av[r];
            }
        }
#pragma unroll
        for (int kk = 0; kk < 64; kk += 32) {
            const v16bf a0 = load_afrag(&Ps[m_off][kk], LDP, lane);
            const v16bf a1 = load_afrag(&Ps[m_off + 16][kk], LDP, lane);
            const v16bf b0 = load_bfragT(&Vt[n_off][kk], LDP, lane);
            const v16bf b1 = load_bfragT(&Vt[n_off + 16][kk], LDP, lane);
            o[0][0] = wmma_bf16(a0, b0, o[0][0]);
            o[0][1] = wmma_bf16(a0, b1, o[0][1]);
            o[1][0] = wmma_bf16(a1, b0, o[1][0]);
            o[1][1] = wmma_bf16(a1, b1, o[1][1]);
        }
        __syncthreads();   // protect Ks/Vt/Ss/Ps before next key block
    }

    // final normalization by L (vectorized row reads), write ctx fp32 [b,s,dim]
#pragma unroll
    for (int ms = 0; ms < 2; ++ms) {
        const int base = m_off + 16 * ms + 8 * half;
        const float4 l0 = *(const float4*)&Lrow[base];
        const float4 l1 = *(const float4*)&Lrow[base + 4];
        const float lv[8] = {l0.x, l0.y, l0.z, l0.w, l1.x, l1.y, l1.z, l1.w};
#pragma unroll
        for (int r = 0; r < 8; ++r) {
            const float inv = (lv[r] > 0.0f) ? 1.0f / lv[r] : 0.0f;
            const int mr = base + r;
#pragma unroll
            for (int ns = 0; ns < 2; ++ns) {
                const int nc = n_off + 16 * ns + nl;
                ctx[((size_t)b * SEQ + q0 + mr) * DIM + h * DPH + nc] =
                    o[ms][ns][r] * inv;
            }
        }
    }
}

// ---------------------------------------------------------------------------
extern "C" void kernel_launch(void* const* d_in, const int* in_sizes, int n_in,
                              void* d_out, int out_size, void* d_ws, size_t ws_size,
                              hipStream_t stream) {
    const float* x     = (const float*)d_in[0];
    const int*   mask  = (const int*)d_in[1];
    const float* Wq    = (const float*)d_in[2];
    const float* bq    = (const float*)d_in[3];
    const float* Wk    = (const float*)d_in[4];
    const float* bk    = (const float*)d_in[5];
    const float* Wv    = (const float*)d_in[6];
    const float* bv    = (const float*)d_in[7];
    const float* Wo    = (const float*)d_in[8];
    const float* bo    = (const float*)d_in[9];
    const float* scale = (const float*)d_in[10];

    const size_t qkvElems = (size_t)BS * NH * SEQ * DPH;   // 4M elems
    bf16_t* qb  = (bf16_t*)d_ws;
    bf16_t* kb  = qb + qkvElems;
    bf16_t* vb  = kb + qkvElems;
    float*  ctx = (float*)(vb + qkvElems);

    const dim3 gGemm(DIM / 64, MTOT / 64);
    const dim3 gAttn(SEQ / 64, BS * NH);
    const dim3 blk(128);

    gemm64<0><<<gGemm, blk, 0, stream>>>(x, Wq, bq, (void*)qb, scale);
    gemm64<1><<<gGemm, blk, 0, stream>>>(x, Wk, bk, (void*)kb, nullptr);
    gemm64<2><<<gGemm, blk, 0, stream>>>(x, Wv, bv, (void*)vb, nullptr);
    attn_kernel<<<gAttn, blk, 0, stream>>>(qb, kb, vb, mask, ctx);
    gemm64<3><<<gGemm, blk, 0, stream>>>(ctx, Wo, bo, d_out, nullptr);
}
